// PointRend_18932215841404
// MI455X (gfx1250) — matrix operand order repaired and checked by reading, not compiled, not run on hardware
//
#include <hip/hip_runtime.h>
#include <hip/hip_bf16.h>

typedef __attribute__((ext_vector_type(16))) _Float16 v16h;
typedef __attribute__((ext_vector_type(8)))  _Float16 v8h;
typedef __attribute__((ext_vector_type(8)))  float    v8f;

namespace {
constexpr int Hh = 512, Wd = 512;
constexpr int CHh = 128, CWw = 128;
constexpr int FHh = 256, FWw = 256;
constexpr int CF = 256, NC = 3;
constexpr int NPTS = 2048;
constexpr int NUM_OVER = 3 * NPTS;   // 6144
constexpr int STEP1 = 1536;          // int(2048*0.75)
constexpr int STEP2 = NPTS - STEP1;  // 512
constexpr int NB = 2;
constexpr int C1 = CF + NC;          // 259 = layer1/layer2 input channels
constexpr int PAD = 288;             // 9 * 32 : K padded for 16x16x32 WMMA
constexpr int KSTEPS = PAD / 32;     // 9
constexpr int C3 = 132;              // 128 + 3 tail + 1 pad
constexpr int PT_TILES = NPTS / 16;  // 128
}

// ---- half-pixel-center bilinear with edge clamp (jax.image.resize upsample) ----
__device__ __forceinline__ float bilin(const float* __restrict__ img, int sh, int sw,
                                       int y, int x, float scale) {
  float sy = ((float)y + 0.5f) * scale - 0.5f;
  float sx = ((float)x + 0.5f) * scale - 0.5f;
  float fy0 = floorf(sy), fx0 = floorf(sx);
  int y0 = (int)fy0, x0 = (int)fx0;
  float wy = sy - fy0, wx = sx - fx0;
  int y0c = min(max(y0, 0), sh - 1);
  int y1c = min(max(y0 + 1, 0), sh - 1);
  int x0c = min(max(x0, 0), sw - 1);
  int x1c = min(max(x0 + 1, 0), sw - 1);
  float v00 = img[y0c * sw + x0c];
  float v01 = img[y0c * sw + x1c];
  float v10 = img[y1c * sw + x0c];
  float v11 = img[y1c * sw + x1c];
  float top = v00 + wx * (v01 - v00);
  float bot = v10 + wx * (v11 - v10);
  return top + wy * (bot - top);
}

// ---- 1) uncertainty of the 6144 candidate points (softmax top2 gap) ----
__global__ void k_uncert(const float* __restrict__ coarse, const int* __restrict__ rnd,
                         float* __restrict__ unc) {
  int t = blockIdx.x * blockDim.x + threadIdx.x;
  if (t >= NB * NUM_OVER) return;
  int b = t / NUM_OVER, i = t % NUM_OVER;
  int lin = rnd[i];
  int y = lin / Wd, x = lin % Wd;
  float v0 = bilin(coarse + ((size_t)b * NC + 0) * (CHh * CWw), CHh, CWw, y, x, 0.25f);
  float v1 = bilin(coarse + ((size_t)b * NC + 1) * (CHh * CWw), CHh, CWw, y, x, 0.25f);
  float v2 = bilin(coarse + ((size_t)b * NC + 2) * (CHh * CWw), CHh, CWw, y, x, 0.25f);
  float m = fmaxf(v0, fmaxf(v1, v2));
  float e0 = expf(v0 - m), e1 = expf(v1 - m), e2 = expf(v2 - m);
  float s = e0 + e1 + e2;
  float p0 = e0 / s, p1 = e1 / s, p2 = e2 / s;
  float first = fmaxf(p0, fmaxf(p1, p2));
  float second = (first == p0) ? fmaxf(p1, p2)
               : (first == p1) ? fmaxf(p0, p2)
                               : fmaxf(p0, p1);
  unc[t] = 1.0f - (first - second);
}

// ---- 2) stable top-k membership via rank (ties -> smaller index first) ----
__global__ void k_selflag(const float* __restrict__ unc, int* __restrict__ flags) {
  int t = blockIdx.x * blockDim.x + threadIdx.x;
  if (t >= NB * NUM_OVER) return;
  int b = t / NUM_OVER, i = t % NUM_OVER;
  const float* u = unc + (size_t)b * NUM_OVER;
  float ui = u[i];
  int rank = 0;
  for (int j = 0; j < NUM_OVER; ++j) {
    float uj = u[j];
    rank += (uj > ui) || (uj == ui && j < i);
  }
  flags[t] = (rank < STEP1) ? 1 : 0;
}

// ---- 3) deterministic compaction of selected candidate lins ----
__global__ void k_compact(const int* __restrict__ flags, const int* __restrict__ rnd,
                          int* __restrict__ pts) {
  int t = blockIdx.x * blockDim.x + threadIdx.x;
  if (t >= NB * NUM_OVER) return;
  int b = t / NUM_OVER, i = t % NUM_OVER;
  const int* f = flags + (size_t)b * NUM_OVER;
  if (!f[i]) return;
  int pos = 0;
  for (int j = 0; j < i; ++j) pos += f[j];
  pts[b * NPTS + pos] = rnd[i];
}

__global__ void k_covfill(const int* __restrict__ rnd, int* __restrict__ pts) {
  int t = blockIdx.x * blockDim.x + threadIdx.x;
  if (t >= NB * STEP2) return;
  int b = t / STEP2, j = t % STEP2;
  pts[b * NPTS + STEP1 + j] = rnd[Hh * Wd - STEP2 + j];
}

// ---- 4) sort 2048 distinct lins per batch by rank-count ----
__global__ void k_sort(const int* __restrict__ pts, int* __restrict__ sorted) {
  int t = blockIdx.x * blockDim.x + threadIdx.x;
  if (t >= NB * NPTS) return;
  int b = t / NPTS, i = t % NPTS;
  const int* p = pts + (size_t)b * NPTS;
  int key = p[i];
  int pos = 0;
  for (int j = 0; j < NPTS; ++j) pos += (p[j] < key);
  sorted[b * NPTS + pos] = key;
}

// ---- 5) gather features (f16 store), replicating the garbled flat reshape ----
// Also side-writes the f32 tail channels (c=256..258) for layer2/layer3 fidelity.
__global__ void k_gather(const float* __restrict__ fine, const float* __restrict__ coarse,
                         const int* __restrict__ sorted, _Float16* __restrict__ featbufh,
                         float* __restrict__ tailf) {
  int t = blockIdx.x * blockDim.x + threadIdx.x;
  if (t >= NB * NPTS * PAD) return;
  int c = t % PAD;
  int bn = t / PAD;
  int n = bn % NPTS, b = bn / NPTS;
  float val = 0.0f;
  if (c < C1) {
    long long f = (long long)b * ((long long)NPTS * C1) + (long long)n * C1 + c;
    const long long FINE_TOTAL = (long long)NB * NPTS * CF;  // 1048576
    if (f < FINE_TOTAL) {
      int bf  = (int)(f / ((long long)NPTS * CF));
      int rem = (int)(f % ((long long)NPTS * CF));
      int nf = rem / CF, cf = rem % CF;
      int lin = sorted[bf * NPTS + nf];
      int y = lin / Wd, x = lin % Wd;
      val = bilin(fine + ((size_t)bf * CF + cf) * (FHh * FWw), FHh, FWw, y, x, 0.5f);
    } else {
      int g = (int)(f - FINE_TOTAL);
      int bc = g / (NPTS * NC);
      int r2 = g % (NPTS * NC);
      int nc2 = r2 / NC, cc = r2 % NC;
      int lin = sorted[bc * NPTS + nc2];
      int y = lin / Wd, x = lin % Wd;
      val = bilin(coarse + ((size_t)bc * NC + cc) * (CHh * CWw), CHh, CWw, y, x, 0.25f);
    }
  }
  featbufh[(size_t)bn * PAD + c] = (_Float16)val;
  if (c >= CF && c < C1) tailf[(size_t)bn * NC + (c - CF)] = val;
}

// ---- 6) pre-convert weights to zero-padded f16 (stride PAD) ----
__global__ void k_wconv(const float* __restrict__ w, _Float16* __restrict__ wh,
                        int outC, int Kvalid) {
  int t = blockIdx.x * blockDim.x + threadIdx.x;
  if (t >= outC * PAD) return;
  int oc = t / PAD, k = t % PAD;
  wh[t] = (k < Kvalid) ? (_Float16)w[(size_t)oc * Kvalid + k] : (_Float16)0.0f;
}

// ---- 7) tail + zero pads for layer2 (f16) / layer3 (f32) inputs ----
__global__ void k_tails(const float* __restrict__ tailf, _Float16* __restrict__ buf2h,
                        float* __restrict__ buf3) {
  int t = blockIdx.x * blockDim.x + threadIdx.x;
  if (t >= NB * NPTS) return;
  const float* tf = tailf + (size_t)t * NC;
  _Float16* b2 = buf2h + (size_t)t * PAD;
  float* b3 = buf3 + (size_t)t * C3;
  for (int j = 0; j < 3; ++j) {
    float v = tf[j];
    b2[CF + j] = (_Float16)v;  // channels 256..258 of layer2 input = tail
    b3[128 + j] = v;           // channels 128..130 of layer3 input = tail
  }
  for (int c = C1; c < PAD; ++c) b2[c] = (_Float16)0.0f;  // zero K-pad
  b3[131] = 0.0f;
}

// ---- 8) WMMA MLP layer: dst = prelu(src[16pts x K] x W^T) per 16x16 tile ----
template <typename DstT>
__global__ void k_mlp_wmma(const _Float16* __restrict__ src, int srcStride,
                           const _Float16* __restrict__ wmat,
                           DstT* __restrict__ dst, int dstStride, int outTiles,
                           const float* __restrict__ aPtr, int totalWaves) {
  int gt = blockIdx.x * blockDim.x + threadIdx.x;
  int wid = gt >> 5;
  if (wid >= totalWaves) return;
  int lane = gt & 31;
  int oc_t = wid % outTiles;
  int tmp = wid / outTiles;
  int pt_t = tmp % PT_TILES;
  int b = tmp / PT_TILES;

  int m = lane & 15;       // row of A within the 16-point tile
  int khalf = lane >> 4;   // lane half selects K sub-block
  int n = lane & 15;       // output channel within tile
  int oc = oc_t * 16 + n;

  const _Float16* srow = src + ((size_t)(b * NPTS) + (size_t)pt_t * 16 + m) * srcStride;
  const _Float16* wrow = wmat + (size_t)oc * PAD;

  v8f acc = {0.f, 0.f, 0.f, 0.f, 0.f, 0.f, 0.f, 0.f};
#pragma unroll
  for (int ks = 0; ks < KSTEPS; ++ks) {
    int kbase = ks * 32;
    // A (16x32 f16): lane half 0 -> K = 0..15, half 1 -> K = 16..31
    v16h A = *(const v16h*)(srow + kbase + khalf * 16);
    // B (32x16 f16): lanes 0-15 hold K=0..15, lanes 16-31 hold K=16..31
    v16h Bv = *(const v16h*)(wrow + kbase + khalf * 16);
    acc = __builtin_amdgcn_wmma_f32_16x16x32_f16(
        /*neg_a=*/false, A, /*neg_b=*/false, Bv,
        /*c_mod=*/(short)0, acc, /*reuse_a=*/false, /*reuse_b=*/false);
  }
  float a = *aPtr;
  // D layout: VGPR r -> M = r + 8*khalf, N = lane&15
#pragma unroll
  for (int r = 0; r < 8; ++r) {
    int M = r + 8 * khalf;
    float v = acc[r];
    v = (v >= 0.0f) ? v : a * v;
    dst[((size_t)(b * NPTS) + (size_t)pt_t * 16 + M) * dstStride + oc] = (DstT)v;
  }
}

// ---- 9) final 3-channel layer (tiny, fp32 VALU): out[b,o,n] ----
__global__ void k_layer3(const float* __restrict__ buf3, const float* __restrict__ w3,
                         float* __restrict__ out) {
  int t = blockIdx.x * blockDim.x + threadIdx.x;
  if (t >= NB * NPTS) return;
  int b = t / NPTS, n = t % NPTS;
  const float* x = buf3 + (size_t)t * C3;
  float a0 = 0.f, a1 = 0.f, a2 = 0.f;
  for (int c = 0; c < 131; ++c) {
    float v = x[c];
    a0 += v * w3[0 * 131 + c];
    a1 += v * w3[1 * 131 + c];
    a2 += v * w3[2 * 131 + c];
  }
  out[(size_t)b * 3 * NPTS + 0 * NPTS + n] = a0;
  out[(size_t)b * 3 * NPTS + 1 * NPTS + n] = a1;
  out[(size_t)b * 3 * NPTS + 2 * NPTS + n] = a2;
}

// ---- 10) mask ----
__global__ void k_maskzero(int* __restrict__ mask) {
  int t = blockIdx.x * blockDim.x + threadIdx.x;
  if (t < NB * Hh * Wd) mask[t] = 0;
}
__global__ void k_maskset(const int* __restrict__ sorted, int* __restrict__ mask) {
  int t = blockIdx.x * blockDim.x + threadIdx.x;
  if (t >= NB * NPTS) return;
  int b = t / NPTS;
  int lin = sorted[t];
  mask[(size_t)b * Hh * Wd + lin] = 1;
}

extern "C" void kernel_launch(void* const* d_in, const int* in_sizes, int n_in,
                              void* d_out, int out_size, void* d_ws, size_t ws_size,
                              hipStream_t stream) {
  const float* fine   = (const float*)d_in[0];  // (2,256,256,256)
  const float* coarse = (const float*)d_in[1];  // (2,3,128,128)
  const float* w1     = (const float*)d_in[2];  // (256,259)
  const float* w2     = (const float*)d_in[3];  // (128,259)
  const float* w3     = (const float*)d_in[4];  // (3,131)
  const float* aPtr   = (const float*)d_in[5];  // prelu_a scalar
  const int*   rnd    = (const int*)d_in[6];    // permutation of 512*512

  char* ws = (char*)d_ws;
  float*     unc      = (float*)ws;     ws += (size_t)NB * NUM_OVER * sizeof(float);
  int*       flags    = (int*)ws;       ws += (size_t)NB * NUM_OVER * sizeof(int);
  int*       pts      = (int*)ws;       ws += (size_t)NB * NPTS * sizeof(int);
  int*       sorted   = (int*)ws;       ws += (size_t)NB * NPTS * sizeof(int);
  _Float16*  featbufh = (_Float16*)ws;  ws += (size_t)NB * NPTS * PAD * sizeof(_Float16);
  _Float16*  buf2h    = (_Float16*)ws;  ws += (size_t)NB * NPTS * PAD * sizeof(_Float16);
  _Float16*  w1h      = (_Float16*)ws;  ws += (size_t)256 * PAD * sizeof(_Float16);
  _Float16*  w2h      = (_Float16*)ws;  ws += (size_t)128 * PAD * sizeof(_Float16);
  float*     tailf    = (float*)ws;     ws += (size_t)NB * NPTS * NC * sizeof(float);
  float*     buf3     = (float*)ws;     ws += (size_t)NB * NPTS * C3 * sizeof(float);

  float* out_mlp = (float*)d_out;                       // (2,3,2048) f32
  int*   mask    = (int*)d_out + (size_t)NB * 3 * NPTS; // (2,512,512) i32

  const int T = 256;
  // weight pre-conversion (independent of point selection)
  k_wconv<<<(256 * PAD + T - 1) / T, T, 0, stream>>>(w1, w1h, 256, C1);
  k_wconv<<<(128 * PAD + T - 1) / T, T, 0, stream>>>(w2, w2h, 128, C1);
  // 1) candidate uncertainty
  k_uncert<<<(NB * NUM_OVER + T - 1) / T, T, 0, stream>>>(coarse, rnd, unc);
  // 2) stable top-1536 membership
  k_selflag<<<(NB * NUM_OVER + T - 1) / T, T, 0, stream>>>(unc, flags);
  // 3) compact + coverage points
  k_compact<<<(NB * NUM_OVER + T - 1) / T, T, 0, stream>>>(flags, rnd, pts);
  k_covfill<<<(NB * STEP2 + T - 1) / T, T, 0, stream>>>(rnd, pts);
  // 4) sort lins
  k_sort<<<(NB * NPTS + T - 1) / T, T, 0, stream>>>(pts, sorted);
  // 5) gather features (on-demand bilinear, f16 store + f32 tail side-channel)
  k_gather<<<(NB * NPTS * PAD + T - 1) / T, T, 0, stream>>>(fine, coarse, sorted,
                                                            featbufh, tailf);
  // 6) tails + K-pad zeros for downstream buffers
  k_tails<<<(NB * NPTS + T - 1) / T, T, 0, stream>>>(tailf, buf2h, buf3);
  // 7) MLP layer1: (2048 x 259) x (259 x 256) -> buf2h[0..255], 9 unrolled WMMAs
  {
    int waves = NB * PT_TILES * 16;  // 4096
    k_mlp_wmma<_Float16><<<(waves * 32 + T - 1) / T, T, 0, stream>>>(
        featbufh, PAD, w1h, buf2h, PAD, 16, aPtr, waves);
  }
  // 8) MLP layer2: (2048 x 259) x (259 x 128) -> buf3[0..127] (f32)
  {
    int waves = NB * PT_TILES * 8;   // 2048
    k_mlp_wmma<float><<<(waves * 32 + T - 1) / T, T, 0, stream>>>(
        buf2h, PAD, w2h, buf3, C3, 8, aPtr, waves);
  }
  // 9) layer3 + mask
  k_layer3<<<(NB * NPTS + T - 1) / T, T, 0, stream>>>(buf3, w3, out_mlp);
  k_maskzero<<<(NB * Hh * Wd + T - 1) / T, T, 0, stream>>>(mask);
  k_maskset<<<(NB * NPTS + T - 1) / T, T, 0, stream>>>(sorted, mask);
}
